// MultiGraph_56023553409634
// MI455X (gfx1250) — matrix operand order repaired and checked by reading, not compiled, not run on hardware
//
#include <hip/hip_runtime.h>
#include <stdint.h>

// ---------------------------------------------------------------------------
// Sizes (match reference)
// ---------------------------------------------------------------------------
#define NNODES 4096
#define XD     200
#define XDP    224                  // XD padded to 7 chunks of K=32
#define HDIM   64
#define EEDGES 131072
#define NGRP   16
#define TOPK   16
#define NH     (NNODES * HDIM)      // 262144 elements per domain

#define WIN_FRAG  (7 * 4 * 32 * 16) // packed w_in halves per domain
#define WMSG_FRAG (2 * 4 * 32 * 16) // packed w_msg halves per (layer,key)

typedef __attribute__((ext_vector_type(16))) _Float16 v16h;
typedef __attribute__((ext_vector_type(4)))  _Float16 v4h;
typedef __attribute__((ext_vector_type(8)))  float    v8f;
typedef __attribute__((ext_vector_type(4)))  uint32_t v4u;

struct Frag32 { v4u lo; v4u hi; };   // 32 bytes == one v16h fragment

// ---------------------------------------------------------------------------
// WMMA per-lane layout (CDNA5 ISA 7.12.2, wave32):
// A 16x32 f16: lane m=L&15; halves = dwords [d0..d0+3] and [d0+8..d0+11],
//              d0 = k0/2 + ((L&16)?4:0)  (two aligned b128 runs per lane)
// B 32x16 f16: lanes 0-15 K=0..7,16..23, lanes 16-31 K=8..15,24..31 (pre-packed in ws)
// ---------------------------------------------------------------------------
__device__ __forceinline__ v16h load_half_frag(const uint32_t* p32, int d0) {
    Frag32 f;
    f.lo = *(const v4u*)(p32 + d0);
    f.hi = *(const v4u*)(p32 + d0 + 8);
    return __builtin_bit_cast(v16h, f);
}

__device__ __forceinline__ float hash_uniform(uint32_t x) {
    x ^= x >> 16; x *= 0x7feb352dU;
    x ^= x >> 15; x *= 0x846ca68bU;
    x ^= x >> 16;
    return ((float)(x >> 8) + 0.5f) * (1.0f / 16777216.0f);
}

// ---------------------------------------------------------------------------
// K0a: pack a [K][64] f32 weight matrix into WMMA B-fragment order:
//      dst[((kc*4+nb)*32 + lane)*16 + hidx], zero-padded past K
// ---------------------------------------------------------------------------
__global__ void pack_wfrag_kernel(const float* __restrict__ src,
                                  _Float16* __restrict__ dst, int K, int nkc) {
    const int tid = blockIdx.x * 256 + threadIdx.x;
    const int total = nkc * 4 * 32 * 16;
    if (tid >= total) return;
    const int hidx = tid & 15;
    const int lane = (tid >> 4) & 31;
    const int nb   = (tid >> 9) & 3;
    const int kc   = tid >> 11;
    const int v = hidx >> 1, odd = hidx & 1;
    const int k = kc * 32 + (hidx & 7) + ((hidx & 8) ? 16 : 0) + ((lane & 16) ? 8 : 0);
    const int n = nb * 16 + (lane & 15);
    dst[tid] = (k < K) ? (_Float16)src[k * HDIM + n] : (_Float16)0.0f;
}

// ---------------------------------------------------------------------------
// K0b: x f32 [4096][200] -> zero-padded f16 [4096][224]
// ---------------------------------------------------------------------------
__global__ void cvt_x_kernel(const float* __restrict__ x,
                             _Float16* __restrict__ xh, int n) {
    const int i = blockIdx.x * 256 + threadIdx.x;
    if (i >= n) return;
    const int k = i % XDP;
    const int r = i / XDP;
    xh[i] = (k < XD) ? (_Float16)x[r * XD + k] : (_Float16)0.0f;
}

// ---------------------------------------------------------------------------
// K1: input projection  h0 = relu(xh @ w_in + b_in)
//     one wave / 16-row tile; per K-chunk: load A + 4 B frags, then 4 WMMA
// ---------------------------------------------------------------------------
__global__ __launch_bounds__(32) void input_proj_kernel(
    const _Float16* __restrict__ xh,    // [4096][224] f16 (padded)
    const _Float16* __restrict__ wpk,   // packed fragments [7][4][32][16]
    const float* __restrict__ bias,     // [64]
    float* __restrict__ hout,           // [4096][64]
    _Float16* __restrict__ hout_h) {    // [4096][64]
    const int lane = threadIdx.x;
    const int row0 = blockIdx.x * 16;
    const uint32_t* p32 = (const uint32_t*)(xh + (size_t)(row0 + (lane & 15)) * XDP);
    const v16h* wp = (const v16h*)wpk;
    const int kg2 = (lane & 16) ? 4 : 0;

    v8f c[4] = {};
#pragma unroll
    for (int kc = 0; kc < 7; ++kc) {
        // phase 1: issue all loads for this K-chunk
        const v16h a = load_half_frag(p32, kc * 16 + kg2);
        v16h bfr[4];
#pragma unroll
        for (int nb = 0; nb < 4; ++nb) bfr[nb] = wp[(kc * 4 + nb) * 32 + lane];
        // phase 2: back-to-back WMMAs
#pragma unroll
        for (int nb = 0; nb < 4; ++nb) {
            c[nb] = __builtin_amdgcn_wmma_f32_16x16x32_f16(
                false, a, false, bfr[nb], (short)0, c[nb], false, false);
        }
    }
#pragma unroll
    for (int nb = 0; nb < 4; ++nb) {
        const int n = nb * 16 + (lane & 15);
        const float bn = bias[n];
#pragma unroll
        for (int r = 0; r < 8; ++r) {
            const int m = row0 + r + ((lane & 16) ? 8 : 0);
            const float v = fmaxf(c[nb][r] + bn, 0.0f);
            hout[m * HDIM + n]   = v;
            hout_h[m * HDIM + n] = (_Float16)v;
        }
    }
}

// ---------------------------------------------------------------------------
// K2: message pass  agg[t][dst] += (h[s][src] @ w_msg) * attn
//     one wave / 16-edge tile.  All operands (2 A frags, 8 B frags, dst idx,
//     attn) are preloaded so the 8 WMMAs issue back-to-back after one wait,
//     then the f32 atomic scatter runs with indices already in registers.
// ---------------------------------------------------------------------------
__global__ __launch_bounds__(32) void msg_wmma_kernel(
    const _Float16* __restrict__ hsrc,  // [4096][64] f16 (source domain)
    const _Float16* __restrict__ wpk,   // packed fragments [2][4][32][16]
    const int* __restrict__ ei,         // [2][E] (src row, dst row)
    const float* __restrict__ attn,     // [E] or nullptr
    float* __restrict__ agg) {          // [4096][64] f32 (target domain)
    const int lane = threadIdx.x;
    const int e0 = blockIdx.x * 16;
    __builtin_prefetch(ei + e0 + 64, 0, 1);            // global_prefetch_b8

    const int srow = ei[e0 + (lane & 15)];
    const uint32_t* p32 = (const uint32_t*)(hsrc + (size_t)srow * HDIM);
    const v16h* wp = (const v16h*)wpk;
    const int kg2 = (lane & 16) ? 4 : 0;

    // ---- phase 1: issue every load up front (A, B, dst rows, attn) ----
    v16h afr[2], bfr[8];
#pragma unroll
    for (int kc = 0; kc < 2; ++kc) afr[kc] = load_half_frag(p32, kc * 16 + kg2);
#pragma unroll
    for (int f = 0; f < 8; ++f) bfr[f] = wp[f * 32 + lane];

    int   drow[8];
    float scale[8];
#pragma unroll
    for (int r = 0; r < 8; ++r) {
        const int er = e0 + r + ((lane & 16) ? 8 : 0);
        drow[r]  = ei[EEDGES + er];
        scale[r] = attn ? attn[er] : 1.0f;
    }

    // ---- phase 2: 8 back-to-back WMMAs ----
    v8f c[4] = {};
#pragma unroll
    for (int kc = 0; kc < 2; ++kc) {
#pragma unroll
        for (int nb = 0; nb < 4; ++nb) {
            c[nb] = __builtin_amdgcn_wmma_f32_16x16x32_f16(
                false, afr[kc], false, bfr[kc * 4 + nb], (short)0,
                c[nb], false, false);
        }
    }

    // ---- phase 3: scaled atomic scatter ----
#pragma unroll
    for (int r = 0; r < 8; ++r) {
        float* arow = agg + (size_t)drow[r] * HDIM;
#pragma unroll
        for (int nb = 0; nb < 4; ++nb) {
            const int n = nb * 16 + (lane & 15);
            atomicAdd(arow + n, c[nb][r] * scale[r]);
        }
    }
}

// ---------------------------------------------------------------------------
// K3: h = relu(h + agg) on float4 granularity; refresh f16 shadow (b64 store)
// ---------------------------------------------------------------------------
__global__ void relu_update_kernel(float* __restrict__ h,
                                   const float* __restrict__ agg,
                                   _Float16* __restrict__ hh, int n4) {
    const int i = blockIdx.x * blockDim.x + threadIdx.x;
    if (i >= n4) return;
    const float4 a = ((const float4*)agg)[i];
    float4 x = ((float4*)h)[i];
    x.x = fmaxf(x.x + a.x, 0.0f);
    x.y = fmaxf(x.y + a.y, 0.0f);
    x.z = fmaxf(x.z + a.z, 0.0f);
    x.w = fmaxf(x.w + a.w, 0.0f);
    ((float4*)h)[i] = x;
    v4h hv;
    hv[0] = (_Float16)x.x; hv[1] = (_Float16)x.y;
    hv[2] = (_Float16)x.z; hv[3] = (_Float16)x.w;
    ((v4h*)hh)[i] = hv;
}

// ---------------------------------------------------------------------------
// K4: edge prediction -> dense N x N scatter-add
// ---------------------------------------------------------------------------
__global__ __launch_bounds__(256) void edge_pred_kernel(
    const float* __restrict__ hs, const float* __restrict__ ht,
    const int* __restrict__ ei,
    const float* __restrict__ wpred_k,  // [128][2] (use column 0)
    const float* __restrict__ bpred_k,  // [2]
    float* __restrict__ dense) {        // [4096][4096]
    __shared__ float wcol[2 * HDIM];
    __shared__ float bias;
    const int tid = threadIdx.x;
    if (tid < 2 * HDIM) wcol[tid] = wpred_k[tid * 2];
    if (tid == 0) bias = bpred_k[0];
    __syncthreads();

    const int e = blockIdx.x * 256 + tid;
    const int i0 = ei[e];
    const int i1 = ei[EEDGES + e];
    const float* f0 = hs + (size_t)i0 * HDIM;
    const float* f1 = ht + (size_t)i1 * HDIM;
    float acc = bias;
#pragma unroll 4
    for (int i = 0; i < HDIM; ++i) acc += f0[i] * wcol[i];
#pragma unroll 4
    for (int i = 0; i < HDIM; ++i) acc += f1[i] * wcol[HDIM + i];
    atomicAdd(&dense[(size_t)i0 * NNODES + i1], acc);
}

// ---------------------------------------------------------------------------
// K5: per-row masked gumbel + hard top-16 selection (row resident in LDS)
// ---------------------------------------------------------------------------
__global__ __launch_bounds__(256) void gumbel_topk_kernel(
    const float* __restrict__ dense, int* __restrict__ topk, int ki) {
    __shared__ float row[NNODES];
    __shared__ float smax[256];
    __shared__ int   simx[256];
    const int tid = threadIdx.x;
    const int r = blockIdx.x;

    for (int i = tid; i < NNODES; i += 256) {
        float v = dense[(size_t)r * NNODES + i];
        float lg = -1e9f;
        if (v > 0.0f) {
            uint32_t h = (uint32_t)(r * NNODES + i) * 0x9E3779B9u ^
                         ((uint32_t)ki + 1u) * 0x85EBCA6Bu;
            float u = fminf(fmaxf(hash_uniform(h), 1e-6f), 1.0f - 1e-6f);
            lg = v - logf(-logf(u));      // logits = dense + gumbel
        }
        row[i] = lg;
    }
    __syncthreads();

    for (int j = 0; j < TOPK; ++j) {
        float best = -3.0e38f; int bi = NNODES;
        for (int i = tid; i < NNODES; i += 256) {
            float v = row[i];
            if (v > best || (v == best && i < bi)) { best = v; bi = i; }
        }
        smax[tid] = best; simx[tid] = bi;
        __syncthreads();
        for (int off = 128; off > 0; off >>= 1) {
            if (tid < off) {
                float ov = smax[tid + off]; int oi = simx[tid + off];
                if (ov > smax[tid] || (ov == smax[tid] && oi < simx[tid])) {
                    smax[tid] = ov; simx[tid] = oi;
                }
            }
            __syncthreads();
        }
        if (tid == 0) { topk[r * TOPK + j] = simx[0]; row[simx[0]] = -3.0e38f; }
        __syncthreads();
    }
}

// ---------------------------------------------------------------------------
// K6: ww[e] = hard straight-through value (1 if edge selected, else 0)
// ---------------------------------------------------------------------------
__global__ void ww_from_topk_kernel(const int* __restrict__ ei,
                                    const int* __restrict__ topk,
                                    float* __restrict__ ww) {
    const int e = blockIdx.x * 256 + threadIdx.x;
    if (e >= EEDGES) return;
    const int r = ei[e];
    const int c = ei[EEDGES + e];
    const int* tk = topk + r * TOPK;
    float v = 0.0f;
#pragma unroll
    for (int j = 0; j < TOPK; ++j) v = (tk[j] == c) ? 1.0f : v;
    ww[e] = v;
}

// ---------------------------------------------------------------------------
// K7: pool row + 64x8 classifier + softmax
// ---------------------------------------------------------------------------
__global__ __launch_bounds__(64) void pool_cls_kernel(
    const float* __restrict__ hdom1, const int* __restrict__ index,
    const float* __restrict__ w_cls, const float* __restrict__ b_cls,
    float* __restrict__ out_yhat, float* __restrict__ out_pool) {
    __shared__ float pool[HDIM];
    __shared__ float logits[8];
    const int t = threadIdx.x;
    const int idx = *index;
    float v = hdom1[(size_t)idx * HDIM + t];
    pool[t] = v;
    out_pool[t] = v;
    __syncthreads();
    if (t < 8) {
        float acc = b_cls[t];
        for (int i = 0; i < HDIM; ++i) acc += pool[i] * w_cls[i * 8 + t];
        logits[t] = acc;
    }
    __syncthreads();
    if (t == 0) {
        float mx = logits[0];
        for (int j = 1; j < 8; ++j) mx = fmaxf(mx, logits[j]);
        float e[8], s = 0.0f;
        for (int j = 0; j < 8; ++j) { e[j] = __expf(logits[j] - mx); s += e[j]; }
        for (int j = 0; j < 8; ++j) out_yhat[j] = e[j] / s;
    }
}

__global__ __launch_bounds__(64) void row_copy_kernel(
    const float* __restrict__ base, const int* __restrict__ index,
    float* __restrict__ out) {
    out[threadIdx.x] = base[(size_t)(*index) * HDIM + threadIdx.x];
}

// ---------------------------------------------------------------------------
// K8: global_add_pool per domain into node_sub [5][16][64]
// ---------------------------------------------------------------------------
__global__ void segpool_kernel(const float* __restrict__ h,
                               const int* __restrict__ b0, const int* __restrict__ b1,
                               const int* __restrict__ b2, const int* __restrict__ b3,
                               const int* __restrict__ b4,
                               float* __restrict__ out) {
    const int gid = blockIdx.x * 256 + threadIdx.x;
    if (gid >= 5 * NH) return;
    const int hcol = gid & (HDIM - 1);
    const int node = (gid >> 6) & (NNODES - 1);
    const int d = gid >> 18;
    const int* b = (d == 0) ? b0 : (d == 1) ? b1 : (d == 2) ? b2 : (d == 3) ? b3 : b4;
    const int g = b[node];
    atomicAdd(out + ((d * NGRP + g) * HDIM + hcol), h[gid]);
}

// ---------------------------------------------------------------------------
// Host orchestration
// ---------------------------------------------------------------------------
static const int ESRC[10] = {0, 1, 2, 3, 1, 0, 1, 2, 1, 3};
static const int EDST[10] = {0, 1, 2, 3, 0, 1, 2, 1, 3, 1};

extern "C" void kernel_launch(void* const* d_in, const int* in_sizes, int n_in,
                              void* d_out, int out_size, void* d_ws, size_t ws_size,
                              hipStream_t stream) {
    (void)in_sizes; (void)n_in; (void)out_size; (void)ws_size;

    const float* x[5];   for (int d = 0; d < 5; ++d) x[d]  = (const float*)d_in[d];
    const int*   ei[10]; for (int k = 0; k < 10; ++k) ei[k] = (const int*)d_in[5 + k];
    const int*   b[5];   for (int d = 0; d < 5; ++d) b[d]  = (const int*)d_in[15 + d];
    const int*   index  = (const int*)d_in[20];
    const float* w_in   = (const float*)d_in[21];
    const float* b_in   = (const float*)d_in[22];
    const float* w_msg  = (const float*)d_in[23];
    const float* w_pred = (const float*)d_in[24];
    const float* b_pred = (const float*)d_in[25];
    const float* w_cls  = (const float*)d_in[26];
    const float* b_cls  = (const float*)d_in[27];

    // output: y_hat(8) | w_weight(4*E) | pool(64) | src_pool(64) | node_sub(5*16*64)
    float* out = (float*)d_out;
    float* out_yhat    = out;
    float* out_ww      = out + 8;
    float* out_pool    = out + 8 + 4 * EEDGES;
    float* out_srcpool = out_pool + HDIM;
    float* out_nodesub = out_srcpool + HDIM;

    char* wsb = (char*)d_ws;
    size_t off = 0;
    auto carve = [&](size_t bytes) -> char* {
        char* p = wsb + off;
        off += (bytes + 255) & ~(size_t)255;
        return p;
    };
    float*    hcur    = (float*)carve(5 * NH * sizeof(float));
    float*    h0      = (float*)carve(5 * NH * sizeof(float));
    float*    h1      = (float*)carve(5 * NH * sizeof(float));
    float*    agg     = (float*)carve(5 * NH * sizeof(float));
    _Float16* hcur_h  = (_Float16*)carve(5 * NH * sizeof(_Float16));
    _Float16* h0_h    = (_Float16*)carve(5 * NH * sizeof(_Float16));
    _Float16* xh      = (_Float16*)carve((size_t)5 * NNODES * XDP * sizeof(_Float16));
    _Float16* w_in_pk  = (_Float16*)carve((size_t)5 * WIN_FRAG * sizeof(_Float16));
    _Float16* w_msg_pk = (_Float16*)carve((size_t)20 * WMSG_FRAG * sizeof(_Float16));
    int*      topk    = (int*)carve(NNODES * TOPK * sizeof(int));
    float*    dense   = (float*)carve((size_t)NNODES * NNODES * sizeof(float));

    // ---- weight packing + x conversion + input projections (shared) ----
    for (int d = 0; d < 5; ++d) {
        pack_wfrag_kernel<<<(WIN_FRAG + 255) / 256, 256, 0, stream>>>(
            w_in + (size_t)d * XD * HDIM, w_in_pk + (size_t)d * WIN_FRAG, XD, 7);
        cvt_x_kernel<<<(NNODES * XDP + 255) / 256, 256, 0, stream>>>(
            x[d], xh + (size_t)d * NNODES * XDP, NNODES * XDP);
    }
    for (int l = 0; l < 2; ++l)
        for (int ki = 0; ki < 10; ++ki)
            pack_wfrag_kernel<<<(WMSG_FRAG + 255) / 256, 256, 0, stream>>>(
                w_msg + (size_t)(l * 10 + ki) * HDIM * HDIM,
                w_msg_pk + (size_t)(l * 10 + ki) * WMSG_FRAG, HDIM, 2);
    for (int d = 0; d < 5; ++d) {
        input_proj_kernel<<<NNODES / 16, 32, 0, stream>>>(
            xh + (size_t)d * NNODES * XDP, w_in_pk + (size_t)d * WIN_FRAG,
            b_in + d * HDIM, h0 + (size_t)d * NH, h0_h + (size_t)d * NH);
    }

    auto run_gnn = [&](bool single, const float* attn_base) {
        hipMemcpyAsync(hcur,   h0,   5 * NH * sizeof(float),    hipMemcpyDeviceToDevice, stream);
        hipMemcpyAsync(hcur_h, h0_h, 5 * NH * sizeof(_Float16), hipMemcpyDeviceToDevice, stream);
        for (int l = 0; l < 2; ++l) {
            hipMemsetAsync(agg, 0, 5 * NH * sizeof(float), stream);
            const int nkeys = single ? 4 : 10;
            for (int ki = 0; ki < nkeys; ++ki) {
                const int s = ESRC[ki], t = EDST[ki];
                const float* attn_p = (attn_base && ki < 4) ? (attn_base + (size_t)ki * EEDGES)
                                                            : nullptr;
                msg_wmma_kernel<<<EEDGES / 16, 32, 0, stream>>>(
                    hcur_h + (size_t)s * NH,
                    w_msg_pk + (size_t)(l * 10 + ki) * WMSG_FRAG,
                    ei[ki], attn_p, agg + (size_t)t * NH);
            }
            relu_update_kernel<<<(5 * NH / 4 + 255) / 256, 256, 0, stream>>>(
                hcur, agg, hcur_h, 5 * NH / 4);
        }
    };

    // 1) single-domain embedding (h1) for edge prediction
    run_gnn(true, nullptr);
    hipMemcpyAsync(h1, hcur, 5 * NH * sizeof(float), hipMemcpyDeviceToDevice, stream);

    // 2) edge sparsification: dense scatter + masked gumbel hard top-K
    for (int ki = 0; ki < 4; ++ki) {
        const int s = ESRC[ki], t = EDST[ki];
        hipMemsetAsync(dense, 0, (size_t)NNODES * NNODES * sizeof(float), stream);
        edge_pred_kernel<<<EEDGES / 256, 256, 0, stream>>>(
            h1 + (size_t)s * NH, h1 + (size_t)t * NH, ei[ki],
            w_pred + (size_t)ki * 2 * HDIM * 2, b_pred + ki * 2, dense);
        gumbel_topk_kernel<<<NNODES, 256, 0, stream>>>(dense, topk, ki);
        ww_from_topk_kernel<<<EEDGES / 256, 256, 0, stream>>>(
            ei[ki], topk, out_ww + (size_t)ki * EEDGES);
    }

    // 3) multi-domain embedding with edge attention -> pool + classify
    run_gnn(false, out_ww);
    pool_cls_kernel<<<1, 64, 0, stream>>>(hcur + (size_t)1 * NH, index,
                                          w_cls, b_cls, out_yhat, out_pool);

    // 4) subgraph single-domain embedding + global_add_pool per domain
    run_gnn(true, out_ww);
    hipMemsetAsync(out_nodesub, 0, 5 * NGRP * HDIM * sizeof(float), stream);
    segpool_kernel<<<(5 * NH + 255) / 256, 256, 0, stream>>>(
        hcur, b[0], b[1], b[2], b[3], b[4], out_nodesub);

    // 5) source multi-domain embedding pool (no attention)
    run_gnn(false, nullptr);
    row_copy_kernel<<<1, 64, 0, stream>>>(hcur + (size_t)1 * NH, index, out_srcpool);
}
